// Attention_30270929502930
// MI455X (gfx1250) — matrix operand order repaired and checked by reading, not compiled, not run on hardware
//
#include <hip/hip_runtime.h>

// Attention pipeline for (N,T,F,E)=(8,64,512,64), f16 WMMA with f32 accumulate.
// K1 : proj   -> Pq/Pk  f16 [n][e][f][t]   (ws)
// K1b: vproj  -> Pv2    f16 [n][e][t][k]   (ws)  (k contiguous: no transpose in K2)
// K2 : attn   -> Ob     f16 [n][e][q][t]   (ws)  async-to-LDS double-buffered K/V
// K3 : oproj  -> out    f32 [n][t][q][d]   (d_out)

typedef _Float16 v16h __attribute__((ext_vector_type(16)));
typedef _Float16 v8h  __attribute__((ext_vector_type(8)));
typedef float    v8f  __attribute__((ext_vector_type(8)));

#define DEVINL __device__ __forceinline__

static constexpr int NB = 8, TT = 64, FF = 512, EE = 64;
static constexpr size_t PROJ_ELEMS = (size_t)NB * EE * FF * TT; // 16,777,216

DEVINL v8f wmma_f16f32(v16h a, v16h b, v8f c) {
  // v_wmma_f32_16x16x32_f16: (neg_a, A, neg_b, B, c_mod, C, reuse_a, reuse_b)
  return __builtin_amdgcn_wmma_f32_16x16x32_f16(false, a, false, b, (short)0, c,
                                                false, false);
}

// A operand (M=16, K=32 f16). `base` is row-major [M][K'] with leading dim ldr.
// ISA layout: lane m(<16): row m, K={0..7,16..23}; lane m+16: row m, K={8..15,24..31}.
DEVINL v16h load_a_frag(const _Float16* base, int ldr, int m0, int k0, int lane) {
  const _Float16* p =
      base + (size_t)(m0 + (lane & 15)) * ldr + k0 + ((lane & 16) ? 8 : 0);
  v8h lo = *(const v8h*)(p);       // K = sel .. sel+7
  v8h hi = *(const v8h*)(p + 16);  // K = sel+16 .. sel+23
  v16h a;
#pragma unroll
  for (int i = 0; i < 8; ++i) { a[i] = lo[i]; a[i + 8] = hi[i]; }
  return a;
}

// B operand (K=32, N=16 f16) sourced from a row-major B^T buffer [N][K'].
// ISA layout: lane n(<16): col n, K={0..7,16..23}; lane n+16: col n, K={8..15,24..31}.
DEVINL v16h load_bt_frag(const _Float16* base, int ldr, int n0, int k0, int lane) {
  const _Float16* p =
      base + (size_t)(n0 + (lane & 15)) * ldr + k0 + ((lane & 16) ? 8 : 0);
  v8h lo = *(const v8h*)(p);
  v8h hi = *(const v8h*)(p + 16);
  v16h b;
#pragma unroll
  for (int i = 0; i < 8; ++i) { b[i] = lo[i]; b[i + 8] = hi[i]; }
  return b;
}

// Issue one 16B async global->LDS copy (ASYNCcnt tracked, no VGPR round-trip).
DEVINL void async_copy_b128(const _Float16* gsrc, _Float16* lds_dst) {
  const unsigned long long ga = (unsigned long long)(uintptr_t)gsrc;
  const unsigned la = (unsigned)(uintptr_t)lds_dst;  // low 32 bits = LDS offset
  asm volatile("global_load_async_to_lds_b128 %0, %1, off"
               :
               : "v"(la), "v"(ga)
               : "memory");
}

// ---------------- K1: Q/K projections ----------------------------------------
// grid (N*F, 2), block 256. blockIdx.y: 0->Q, 1->K.
// For fixed (n,f): D[d][t] = sum_e W[d][e] * X[t][e]; store to P[n][d][f][t].
__global__ __launch_bounds__(256) void proj_kernel(
    const float* __restrict__ key, const float* __restrict__ query,
    const float* __restrict__ Wk, const float* __restrict__ Wq,
    _Float16* __restrict__ Pq, _Float16* __restrict__ Pk) {
  const int m = blockIdx.y;
  const float* X = (m == 0) ? query : key;
  const float* W = (m == 0) ? Wq : Wk;
  _Float16* P = (m == 0) ? Pq : Pk;

  const int n = blockIdx.x / FF;
  const int f = blockIdx.x % FF;
  const int tid = threadIdx.x, wave = tid >> 5, lane = tid & 31;

  __shared__ _Float16 Xs[TT * EE];   // [t][e]
  __shared__ _Float16 Wsh[EE * EE];  // [d][e]

  {
    const int r = tid >> 2, e0 = (tid & 3) * 16;
    const float4* xsrc =
        (const float4*)(X + (((size_t)(n * TT + r) * FF + f) * EE + e0));
    const float4* wsrc = (const float4*)(W + (size_t)r * EE + e0);
#pragma unroll
    for (int i = 0; i < 4; ++i) {
      float4 xv = xsrc[i];
      Xs[r * EE + e0 + i * 4 + 0] = (_Float16)xv.x;
      Xs[r * EE + e0 + i * 4 + 1] = (_Float16)xv.y;
      Xs[r * EE + e0 + i * 4 + 2] = (_Float16)xv.z;
      Xs[r * EE + e0 + i * 4 + 3] = (_Float16)xv.w;
      float4 wv = wsrc[i];
      Wsh[r * EE + e0 + i * 4 + 0] = (_Float16)wv.x;
      Wsh[r * EE + e0 + i * 4 + 1] = (_Float16)wv.y;
      Wsh[r * EE + e0 + i * 4 + 2] = (_Float16)wv.z;
      Wsh[r * EE + e0 + i * 4 + 3] = (_Float16)wv.w;
    }
  }
  __syncthreads();

#pragma unroll
  for (int i = 0; i < 2; ++i) {
    const int id = wave * 2 + i;
    const int mtd = id >> 2;  // d tile
    const int ntt = id & 3;   // t tile
    v8f acc = {};
#pragma unroll
    for (int c = 0; c < 2; ++c) {
      v16h a = load_a_frag(Wsh, EE, mtd * 16, c * 32, lane);  // A: W[d][e]
      v16h b = load_bt_frag(Xs, EE, ntt * 16, c * 32, lane);  // B^T: X[t][e]
      acc = wmma_f16f32(a, b, acc);
    }
#pragma unroll
    for (int j = 0; j < 8; ++j) {
      const int d = mtd * 16 + j + ((lane & 16) ? 8 : 0);
      const int t = ntt * 16 + (lane & 15);
      P[((size_t)(n * EE + d) * FF + f) * TT + t] = (_Float16)acc[j];
    }
  }
}

// ---------------- K1b: V projection, transposed layout -----------------------
// grid N*T*2 (f in halves of 256), block 256. For fixed (n,t):
// D[d][f] = sum_e Wv[d][e] * value[n,t,f,e]; store to Pv2[n][d][t][f].
__global__ __launch_bounds__(256) void vproj_kernel(
    const float* __restrict__ value, const float* __restrict__ Wv,
    _Float16* __restrict__ Pv2) {
  const int bx = blockIdx.x;
  const int n = bx >> 7;
  const int t = (bx >> 1) & 63;
  const int f0 = (bx & 1) * 256;
  const int tid = threadIdx.x, wave = tid >> 5, lane = tid & 31;

  __shared__ _Float16 Xs[256 * EE];  // [f][e]  32 KB
  __shared__ _Float16 Wsh[EE * EE];  // [d][e]   8 KB

  {
    const int r = tid >> 2, e0 = (tid & 3) * 16;
    const float4* wsrc = (const float4*)(Wv + (size_t)r * EE + e0);
#pragma unroll
    for (int i = 0; i < 4; ++i) {
      float4 wv = wsrc[i];
      Wsh[r * EE + e0 + i * 4 + 0] = (_Float16)wv.x;
      Wsh[r * EE + e0 + i * 4 + 1] = (_Float16)wv.y;
      Wsh[r * EE + e0 + i * 4 + 2] = (_Float16)wv.z;
      Wsh[r * EE + e0 + i * 4 + 3] = (_Float16)wv.w;
    }
    // one 64-float row of X per thread
    const float4* xsrc = (const float4*)(
        value + ((size_t)(n * TT + t) * FF + f0 + tid) * EE);
#pragma unroll
    for (int i = 0; i < 16; ++i) {
      float4 xv = xsrc[i];
      Xs[tid * EE + i * 4 + 0] = (_Float16)xv.x;
      Xs[tid * EE + i * 4 + 1] = (_Float16)xv.y;
      Xs[tid * EE + i * 4 + 2] = (_Float16)xv.z;
      Xs[tid * EE + i * 4 + 3] = (_Float16)xv.w;
    }
  }
  __syncthreads();

  // 4 d-tiles x 16 f-tiles = 64 tiles, 8 per wave
#pragma unroll
  for (int i = 0; i < 8; ++i) {
    const int id = wave * 8 + i;
    const int mtd = id >> 4;  // d tile
    const int ntf = id & 15;  // f tile
    v8f acc = {};
#pragma unroll
    for (int c = 0; c < 2; ++c) {
      v16h a = load_a_frag(Wsh, EE, mtd * 16, c * 32, lane);  // A: Wv[d][e]
      v16h b = load_bt_frag(Xs, EE, ntf * 16, c * 32, lane);  // B^T: X[f][e]
      acc = wmma_f16f32(a, b, acc);
    }
#pragma unroll
    for (int j = 0; j < 8; ++j) {
      const int d = mtd * 16 + j + ((lane & 16) ? 8 : 0);
      const int f = f0 + ntf * 16 + (lane & 15);
      Pv2[((size_t)(n * EE + d) * TT + t) * FF + f] = (_Float16)acc[j];
    }
  }
}

// ---------------- K2: attention per (n,e) ------------------------------------
// grid N*E*2 (q split in halves of 256), block 256 (8 waves, 32 q-rows/wave).
// K/V chunks staged by double-buffered global_load_async_to_lds_b128.
__global__ __launch_bounds__(256) void attn_kernel(
    const _Float16* __restrict__ Pq, const _Float16* __restrict__ Pk,
    const _Float16* __restrict__ Pv2, _Float16* __restrict__ Ob) {
  constexpr int NC = FF / 32;  // 16 k-chunks
  const int bx = blockIdx.x;
  const int ne = bx >> 1;
  const int half = bx & 1;
  const int tid = threadIdx.x, wave = tid >> 5, lane = tid & 31;
  const int qwave = half * 256 + wave * 32;

  __shared__ _Float16 KsB[2][32 * TT];    // K chunk [k][t], double-buffered
  __shared__ _Float16 VtB[2][TT * 32];    // V chunk [t][k], double-buffered
  __shared__ _Float16 Ps[8][16 * 32];     // per-wave P tile [q][k]

  const _Float16* Qb = Pq + (size_t)ne * FF * TT;
  const _Float16* Kb = Pk + (size_t)ne * FF * TT;
  const _Float16* Vb = Pv2 + (size_t)ne * TT * FF;  // [t][k] rows

  // per-thread async staging coordinates (8 halves = 16 B per copy)
  const int kr = tid >> 3, kc = (tid & 7) * 8;   // K: 32 rows x 8 segs
  const int vr = tid >> 2, vc = (tid & 3) * 8;   // V: 64 rows x 4 segs

  auto issue_chunk = [&](int kb, int buf) {
    async_copy_b128(Kb + (size_t)(kb * 32 + kr) * TT + kc,
                    &KsB[buf][kr * TT + kc]);
    async_copy_b128(Vb + (size_t)vr * FF + kb * 32 + vc,
                    &VtB[buf][vr * 32 + vc]);
  };

  // Q fragments held in registers: [mtile][t-chunk]
  v16h qf[2][2];
#pragma unroll
  for (int mt = 0; mt < 2; ++mt)
#pragma unroll
    for (int c = 0; c < 2; ++c)
      qf[mt][c] = load_a_frag(Qb, TT, qwave + mt * 16, c * 32, lane);

  v8f Oacc[2][4];
  v8f Lacc[2];
#pragma unroll
  for (int mt = 0; mt < 2; ++mt) {
    Lacc[mt] = (v8f){};
#pragma unroll
    for (int tt = 0; tt < 4; ++tt) Oacc[mt][tt] = (v8f){};
  }

  v16h ones;
#pragma unroll
  for (int i = 0; i < 16; ++i) ones[i] = (_Float16)1.0f;

  const float KSCALE = 0.18033688011112042f;  // log2(e)/sqrt(EMBED)

  issue_chunk(0, 0);

  for (int kb = 0; kb < NC; ++kb) {
    const int cur = kb & 1;
    if (kb + 1 < NC) {
      issue_chunk(kb + 1, cur ^ 1);              // prefetch next chunk
      asm volatile("s_wait_asynccnt 0x2" ::: "memory");  // current chunk done
    } else {
      asm volatile("s_wait_asynccnt 0x0" ::: "memory");
    }
    __syncthreads();

    const _Float16* Ks = KsB[cur];
    const _Float16* Vt = VtB[cur];

    // B fragments of K^T (for S) and V (for O); shared across both q-mtiles.
    v16h kf[2][2];  // [k-subtile][t-chunk]
#pragma unroll
    for (int s = 0; s < 2; ++s)
#pragma unroll
      for (int c = 0; c < 2; ++c)
        kf[s][c] = load_bt_frag(Ks, TT, s * 16, c * 32, lane);
    v16h vf[4];  // [t-tile], K = 32 k's
#pragma unroll
    for (int tt = 0; tt < 4; ++tt)
      vf[tt] = load_bt_frag(Vt, 32, tt * 16, 0, lane);

#pragma unroll
    for (int mt = 0; mt < 2; ++mt) {
      v8f S0 = {}, S1 = {};
      S0 = wmma_f16f32(qf[mt][0], kf[0][0], S0);
      S0 = wmma_f16f32(qf[mt][1], kf[0][1], S0);
      S1 = wmma_f16f32(qf[mt][0], kf[1][0], S1);
      S1 = wmma_f16f32(qf[mt][1], kf[1][1], S1);

      // P = exp(S/8) (logits are O(1): no max subtraction needed), to LDS
      const int qrow = (lane & 16) ? 8 : 0;
      _Float16* pw = Ps[wave];
#pragma unroll
      for (int j = 0; j < 8; ++j) {
        pw[(j + qrow) * 32 + (lane & 15)]      = (_Float16)exp2f(S0[j] * KSCALE);
        pw[(j + qrow) * 32 + 16 + (lane & 15)] = (_Float16)exp2f(S1[j] * KSCALE);
      }
      asm volatile("s_wait_dscnt 0" ::: "memory");  // wave-local LDS round-trip
      v16h pf = load_a_frag(pw, 32, 0, 0, lane);

#pragma unroll
      for (int tt = 0; tt < 4; ++tt)
        Oacc[mt][tt] = wmma_f16f32(pf, vf[tt], Oacc[mt][tt]);
      Lacc[mt] = wmma_f16f32(pf, ones, Lacc[mt]);  // row sums, same C layout
    }
    __syncthreads();  // all waves done with buf `cur` before it is re-filled
  }

  // normalize and store: Ob[n][e][q][t]
  _Float16* Obase = Ob + (size_t)ne * FF * TT;
#pragma unroll
  for (int mt = 0; mt < 2; ++mt) {
#pragma unroll
    for (int j = 0; j < 8; ++j) {
      const float rl = __builtin_amdgcn_rcpf(Lacc[mt][j]);
      const int q = qwave + mt * 16 + j + ((lane & 16) ? 8 : 0);
#pragma unroll
      for (int tt = 0; tt < 4; ++tt)
        Obase[(size_t)q * TT + tt * 16 + (lane & 15)] =
            (_Float16)(Oacc[mt][tt][j] * rl);
    }
  }
}

// ---------------- K3: output projection + bias -------------------------------
// grid N*F, block 256. For fixed (n,q): out[t][d] = sum_e O[t][e]*Wo[d][e]+bo[d]
__global__ __launch_bounds__(256) void oproj_kernel(
    const _Float16* __restrict__ Ob, const float* __restrict__ Wo,
    const float* __restrict__ bo, float* __restrict__ out) {
  const int n = blockIdx.x / FF;
  const int q = blockIdx.x % FF;
  const int tid = threadIdx.x, wave = tid >> 5, lane = tid & 31;

  __shared__ _Float16 Os[TT * EE];   // [t][e]
  __shared__ _Float16 Wsh[EE * EE];  // [d][e]

  {
    const int r = tid >> 2, t0 = (tid & 3) * 16;  // r = e (for Os) and d (for W)
    const _Float16* src = Ob + ((size_t)(n * EE + r) * FF + q) * TT + t0;
    v8h a = *(const v8h*)src;
    v8h b = *(const v8h*)(src + 8);
#pragma unroll
    for (int i = 0; i < 8; ++i) {
      Os[(t0 + i) * EE + r] = a[i];
      Os[(t0 + 8 + i) * EE + r] = b[i];
    }
    const float4* wsrc = (const float4*)(Wo + (size_t)r * EE + t0);
#pragma unroll
    for (int i = 0; i < 4; ++i) {
      float4 wv = wsrc[i];
      Wsh[r * EE + t0 + i * 4 + 0] = (_Float16)wv.x;
      Wsh[r * EE + t0 + i * 4 + 1] = (_Float16)wv.y;
      Wsh[r * EE + t0 + i * 4 + 2] = (_Float16)wv.z;
      Wsh[r * EE + t0 + i * 4 + 3] = (_Float16)wv.w;
    }
  }
  __syncthreads();

#pragma unroll
  for (int i = 0; i < 2; ++i) {
    const int id = wave * 2 + i;
    const int mtt = id >> 2;  // t tile
    const int ntd = id & 3;   // d tile
    v8f acc = {};
#pragma unroll
    for (int c = 0; c < 2; ++c) {
      v16h a = load_a_frag(Os, EE, mtt * 16, c * 32, lane);    // A: O[t][e]
      v16h b = load_bt_frag(Wsh, EE, ntd * 16, c * 32, lane);  // B^T: Wo[d][e]
      acc = wmma_f16f32(a, b, acc);
    }
    const float bias = bo[ntd * 16 + (lane & 15)];
#pragma unroll
    for (int j = 0; j < 8; ++j) {
      const int t = mtt * 16 + j + ((lane & 16) ? 8 : 0);
      out[((size_t)(n * TT + t) * FF + q) * EE + ntd * 16 + (lane & 15)] =
          acc[j] + bias;
    }
  }
}

// -----------------------------------------------------------------------------
extern "C" void kernel_launch(void* const* d_in, const int* in_sizes, int n_in,
                              void* d_out, int out_size, void* d_ws,
                              size_t ws_size, hipStream_t stream) {
  const float* value = (const float*)d_in[0];
  const float* key   = (const float*)d_in[1];
  const float* query = (const float*)d_in[2];
  const float* Wv    = (const float*)d_in[3];
  const float* Wk    = (const float*)d_in[4];
  const float* Wq    = (const float*)d_in[5];
  const float* Wo    = (const float*)d_in[6];
  const float* bo    = (const float*)d_in[7];

  // workspace layout: Pq | Pk | Pv2 | Ob  (4 x 32 MiB f16 = 128 MiB)
  _Float16* Pq  = (_Float16*)d_ws;
  _Float16* Pk  = Pq + PROJ_ELEMS;
  _Float16* Pv2 = Pk + PROJ_ELEMS;
  _Float16* Ob  = Pv2 + PROJ_ELEMS;

  proj_kernel<<<dim3(NB * FF, 2), 256, 0, stream>>>(key, query, Wk, Wq, Pq, Pk);
  vproj_kernel<<<dim3(NB * TT * 2), 256, 0, stream>>>(value, Wv, Pv2);
  attn_kernel<<<dim3(NB * EE * 2), 256, 0, stream>>>(Pq, Pk, Pv2, Ob);
  oproj_kernel<<<dim3(NB * FF), 256, 0, stream>>>(Ob, Wo, bo, (float*)d_out);

  (void)in_sizes; (void)n_in; (void)out_size; (void)ws_size;
}
